// wrap_model_38637525795124
// MI455X (gfx1250) — matrix operand order repaired and checked by reading, not compiled, not run on hardware
//
#include <hip/hip_runtime.h>
#include <hip/hip_bf16.h>
#include <stdint.h>

#define DD 256

typedef __attribute__((ext_vector_type(16))) __bf16 v16bf;
typedef __attribute__((ext_vector_type(8)))  float  v8f;

union FragAB { v16bf v; unsigned int u[8]; };
union FragC  { v8f   v; float f[8]; };

// ---------------- weight prep: transpose + convert to bf16 (Wt[n][k]) ----------------
__global__ void prep_weights(const float* __restrict__ mlpW, const float* __restrict__ nnW,
                             const float* __restrict__ aW,
                             __bf16* __restrict__ W1t, __bf16* __restrict__ W2t,
                             __bf16* __restrict__ W3t, __bf16* __restrict__ nnWt,
                             __bf16* __restrict__ aWt) {
  int id = blockIdx.x * blockDim.x + threadIdx.x;   // 256*256 threads
  int n = id >> 8, k = id & 255;
  W1t[n * 256 + k]  = (__bf16)mlpW[k * 256 + n];
  W2t[n * 256 + k]  = (__bf16)mlpW[(256 + k) * 256 + n];
  W3t[n * 256 + k]  = (__bf16)mlpW[(512 + k) * 256 + n];
  nnWt[n * 256 + k] = (__bf16)nnW[k * 256 + n];
  if (k < 96) aWt[n * 96 + k] = (k < 92) ? (__bf16)aW[k * 256 + n] : (__bf16)0.0f;
}

// ---------------- per-layer rank-1 edge terms: u_l = u@W3, v_l = v@W3 + mlpb --------
__global__ void prep_uv(const float* __restrict__ mlpW, const float* __restrict__ mlpb,
                        const float* __restrict__ eW, const float* __restrict__ eb,
                        float* __restrict__ u, float* __restrict__ v) {
  __shared__ float uc[256], vc[256];
  int d = threadIdx.x;
  uc[d] = eW[d];
  vc[d] = eb[d];
  __syncthreads();
  for (int l = 0; l < 4; ++l) {
    float un = 0.f, vn = 0.f;
    for (int k = 0; k < 256; ++k) {
      float w = mlpW[(512 + k) * 256 + d];  // W3[k][d]
      un += uc[k] * w;
      vn += vc[k] * w;
    }
    vn += mlpb[d];
    __syncthreads();
    uc[d] = un; vc[d] = vn;
    __syncthreads();
    u[l * 256 + d] = un;
    v[l * 256 + d] = vn;
  }
}

// stage one 128x32 tile into LDS buffer.
// bf16 source (row stride 256, padded rows): async 16B copies via TDM-class async op.
// f32 source: synchronous guarded load + convert (completes at the __syncthreads).
__device__ __forceinline__ void stage_tile(const void* src, int isbf16, int m0, int kb,
                                           int M, int sA, int Ktrue, __bf16* buf, int tid) {
  if (isbf16) {
    const __bf16* S = (const __bf16*)src;
#pragma unroll
    for (int j = 0; j < 2; ++j) {
      int c = tid + j * 256;              // 512 chunks of 16B
      int r = c >> 2, q = c & 3;
      const __bf16* g = S + (size_t)(m0 + r) * 256 + kb + q * 8;
      unsigned int lds = (unsigned int)(size_t)(buf + r * 32 + q * 8);
      unsigned long long ga = (unsigned long long)(size_t)g;
      asm volatile("global_load_async_to_lds_b128 %0, %1, off"
                   :: "v"(lds), "v"(ga) : "memory");
    }
  } else {
    const float* S = (const float*)src;
#pragma unroll
    for (int j = 0; j < 16; ++j) {
      int id = tid + j * 256;
      int r = id >> 5, c = id & 31;
      int m = m0 + r, k = kb + c;
      float val = (m < M && k < Ktrue) ? S[(size_t)m * sA + k] : 0.f;
      buf[r * 32 + c] = (__bf16)val;
    }
  }
}

// ---------------- WMMA GEMM: C[M][256](bf16) = A1@W1 (+ A2@W2) (+bias) (+BN+ReLU) ----
__global__ __launch_bounds__(256) void gemm_wmma(
    __bf16* __restrict__ C,
    const void* __restrict__ A1, int t1, const __bf16* __restrict__ Wt1,
    const void* __restrict__ A2, int t2, const __bf16* __restrict__ Wt2,
    int M, int sA, int Kp, int Ktrue,
    const float* __restrict__ bias, int mode,
    const float* __restrict__ bn_g, const float* __restrict__ bn_b,
    const float* __restrict__ bn_m, const float* __restrict__ bn_v) {
  __shared__ __attribute__((aligned(16))) __bf16 As[2][128 * 32];  // 2 x 8 KB

  const int tid  = threadIdx.x;
  const int lane = tid & 31;
  const int wid  = tid >> 5;       // 8 waves
  const int wm   = wid & 3;        // 4 waves in M (32 rows each)
  const int wn   = wid >> 2;       // 2 waves in N (128 cols each)
  const int m0   = blockIdx.x * 128;
  const int hf   = lane >> 4;
  const int ml   = lane & 15;

  v8f acc[2][8];
  v8f zero = {0.f, 0.f, 0.f, 0.f, 0.f, 0.f, 0.f, 0.f};
#pragma unroll
  for (int mi = 0; mi < 2; ++mi)
#pragma unroll
    for (int ni = 0; ni < 8; ++ni) acc[mi][ni] = zero;

  const int nsrc = (A2 != nullptr) ? 2 : 1;
  const int spk  = Kp / 32;
  const int total = nsrc * spk;

  // prologue: stage step 0 into buffer 0
  stage_tile((0 < spk) ? A1 : A2, (0 < spk) ? t1 : t2, m0, 0, M, sA, Ktrue, As[0], tid);
  asm volatile("s_wait_asynccnt 0x0" ::: "memory");
  __syncthreads();

  for (int s = 0; s < total; ++s) {
    const int cur = s & 1;
    if (s + 1 < total) {
      const int sn = s + 1;
      stage_tile((sn < spk) ? A1 : A2, (sn < spk) ? t1 : t2,
                 m0, (sn % spk) * 32, M, sA, Ktrue, As[cur ^ 1], tid);
    }
    const __bf16* Wt = (s < spk) ? Wt1 : Wt2;
    const int kb = (s % spk) * 32;
    const __bf16* buf = As[cur];

    // A fragments, contiguous k = 16*half + s (m = lane&15; K split by lane half)
    FragAB af[2];
#pragma unroll
    for (int mi = 0; mi < 2; ++mi) {
      const __bf16* base = buf + (wm * 32 + mi * 16 + ml) * 32;
#pragma unroll
      for (int i = 0; i < 4; ++i)
        af[mi].u[i] = *(const unsigned int*)(base + hf * 16 + 2 * i);
#pragma unroll
      for (int i = 0; i < 4; ++i)
        af[mi].u[4 + i] = *(const unsigned int*)(base + 8 + hf * 16 + 2 * i);
    }
    // B fragments straight from global bf16 Wt (n = lane&15; K = half*16..+15)
#pragma unroll
    for (int ni = 0; ni < 8; ++ni) {
      FragAB bf;
      const uint4* p = (const uint4*)(Wt + (size_t)(wn * 128 + ni * 16 + ml) * Kp +
                                      kb + hf * 16);
      uint4 q0 = p[0], q1 = p[1];
      bf.u[0] = q0.x; bf.u[1] = q0.y; bf.u[2] = q0.z; bf.u[3] = q0.w;
      bf.u[4] = q1.x; bf.u[5] = q1.y; bf.u[6] = q1.z; bf.u[7] = q1.w;
#pragma unroll
      for (int mi = 0; mi < 2; ++mi)
        acc[mi][ni] = __builtin_amdgcn_wmma_f32_16x16x32_bf16(
            false, af[mi].v, false, bf.v, (short)0, acc[mi][ni], false, false);
    }
    // next tile's async copies must have landed, and everyone must be done
    // reading the current buffer before it is refilled at the start of the next step
    asm volatile("s_wait_asynccnt 0x0" ::: "memory");
    __syncthreads();
  }

  // epilogue: optional bias, optional fused BatchNorm+ReLU, bf16 store
#pragma unroll
  for (int mi = 0; mi < 2; ++mi) {
    int mbase = m0 + wm * 32 + mi * 16 + hf * 8;
#pragma unroll
    for (int ni = 0; ni < 8; ++ni) {
      int n = wn * 128 + ni * 16 + ml;
      FragC r; r.v = acc[mi][ni];
      float b = bias ? bias[n] : 0.f;
      float s = 1.f, t = 0.f;
      if (mode == 1) {
        float rs = rsqrtf(bn_v[n] + 1e-5f);
        s = bn_g[n] * rs;
        t = bn_b[n] - bn_m[n] * s;
      }
#pragma unroll
      for (int i = 0; i < 8; ++i) {
        int m = mbase + i;
        if (m < M) {
          float val = r.f[i] + b;
          if (mode == 1) val = fmaxf(val * s + t, 0.f);
          C[(size_t)m * DD + n] = (__bf16)val;
        }
      }
    }
  }
}

// ---------------- edge message + scatter: agg[col] += relu(x[row]+A[row]+B[col]+aw*u+v)
__global__ void edge_kernel(const int* __restrict__ ei, const float* __restrict__ aw,
                            const __bf16* __restrict__ x, const __bf16* __restrict__ Am,
                            const __bf16* __restrict__ Bm, const float* __restrict__ u,
                            const float* __restrict__ v, float* __restrict__ agg, int E) {
  long long id = (long long)blockIdx.x * 256 + threadIdx.x;
  if (id >= (long long)E * DD) return;
  int e = (int)(id >> 8), d = (int)(id & 255);
  int r = ei[e], c = ei[E + e];
  float m = (float)x[(size_t)r * DD + d] + (float)Am[(size_t)r * DD + d] +
            (float)Bm[(size_t)c * DD + d] + aw[e] * u[d] + v[d];
  m = fmaxf(m, 0.f);
  atomicAdd(&agg[(size_t)c * DD + d], m);
}

// ---------------- graph pooling ----------------
__global__ void pool_kernel(const __bf16* __restrict__ x, const int* __restrict__ batch,
                            float* __restrict__ pooled, int N) {
  size_t id = (size_t)blockIdx.x * 256 + threadIdx.x;
  if (id >= (size_t)N * DD) return;
  int n = (int)(id >> 8), d = (int)(id & 255);
  atomicAdd(&pooled[(size_t)batch[n] * DD + d], (float)x[id]);
}

// ---------------- output head: 1500x256 @ 256x6 ----------------
__global__ void out_kernel(const float* __restrict__ pooled, const float* __restrict__ outW,
                           const float* __restrict__ outb, float* __restrict__ out, int G) {
  int id = blockIdx.x * 256 + threadIdx.x;
  if (id >= G * 6) return;
  int g = id / 6, j = id % 6;
  float s = outb[j];
  for (int k = 0; k < 256; ++k) s += pooled[g * 256 + k] * outW[k * 6 + j];
  out[id] = s;
}

extern "C" void kernel_launch(void* const* d_in, const int* in_sizes, int n_in,
                              void* d_out, int out_size, void* d_ws, size_t ws_size,
                              hipStream_t stream) {
  const float* node_features = (const float*)d_in[0];
  const int*   edge_index    = (const int*)d_in[1];
  const float* atom_weights  = (const float*)d_in[2];
  const int*   batch         = (const int*)d_in[3];
  const float* aW   = (const float*)d_in[4];
  const float* ab   = (const float*)d_in[5];
  const float* eW   = (const float*)d_in[6];
  const float* eb   = (const float*)d_in[7];
  const float* mlpW = (const float*)d_in[8];
  const float* mlpb = (const float*)d_in[9];
  const float* nnW  = (const float*)d_in[10];
  const float* nnb  = (const float*)d_in[11];
  const float* bn_g = (const float*)d_in[12];
  const float* bn_b = (const float*)d_in[13];
  const float* bn_m = (const float*)d_in[14];
  const float* bn_v = (const float*)d_in[15];
  const float* outW = (const float*)d_in[16];
  const float* outb = (const float*)d_in[17];

  const int N = in_sizes[0] / 92;
  const int E = in_sizes[2];
  const int G = out_size / 6;
  const int Npad = (N + 127) & ~127;
  const size_t NDp = (size_t)Npad * DD;
  const size_t ND  = (size_t)N * DD;

  // workspace carve-up: f32 arrays first, then bf16 (all 16B aligned)
  char* p = (char*)d_ws;
  float* agg    = (float*)p; p += NDp * 4;
  float* pooled = (float*)p; p += (size_t)G * 256 * 4;
  float* u      = (float*)p; p += 4 * 256 * 4;
  float* v      = (float*)p; p += 4 * 256 * 4;
  __bf16* x    = (__bf16*)p; p += NDp * 2;
  __bf16* Am   = (__bf16*)p; p += NDp * 2;
  __bf16* Bm   = (__bf16*)p; p += NDp * 2;
  __bf16* W1t  = (__bf16*)p; p += 256 * 256 * 2;
  __bf16* W2t  = (__bf16*)p; p += 256 * 256 * 2;
  __bf16* W3t  = (__bf16*)p; p += 256 * 256 * 2;
  __bf16* nnWt = (__bf16*)p; p += 256 * 256 * 2;
  __bf16* aWt  = (__bf16*)p; p += 256 * 96 * 2;

  prep_weights<<<256, 256, 0, stream>>>(mlpW, nnW, aW, W1t, W2t, W3t, nnWt, aWt);
  prep_uv<<<1, 256, 0, stream>>>(mlpW, mlpb, eW, eb, u, v);

  const int gGrid = Npad / 128;

  // x0 = node_features @ aW + ab   (f32 source, K = 92 padded to 96)
  gemm_wmma<<<gGrid, 256, 0, stream>>>(x, node_features, 0, aWt, nullptr, 0, nullptr,
                                       N, 92, 96, 92, ab, 0,
                                       nullptr, nullptr, nullptr, nullptr);
  (void)hipMemsetAsync(Am, 0, NDp * 2, stream);
  (void)hipMemsetAsync(Bm, 0, NDp * 2, stream);

  const long long edgeTot = (long long)E * DD;
  for (int l = 0; l < 4; ++l) {
    // A <- A@W3 + x@W1 ; B <- B@W3 + x@W2   (bf16 sources -> async-to-LDS staging)
    gemm_wmma<<<gGrid, 256, 0, stream>>>(Am, Am, 1, W3t, x, 1, W1t, N, DD, DD, DD,
                                         nullptr, 0, nullptr, nullptr, nullptr, nullptr);
    gemm_wmma<<<gGrid, 256, 0, stream>>>(Bm, Bm, 1, W3t, x, 1, W2t, N, DD, DD, DD,
                                         nullptr, 0, nullptr, nullptr, nullptr, nullptr);
    (void)hipMemsetAsync(agg, 0, ND * 4, stream);
    edge_kernel<<<(int)((edgeTot + 255) / 256), 256, 0, stream>>>(
        edge_index, atom_weights, x, Am, Bm, u + l * 256, v + l * 256, agg, E);
    // x <- relu(BN((agg + x)@nnW + nnb))  as agg@nnW + x@nnW with fused epilogue
    gemm_wmma<<<gGrid, 256, 0, stream>>>(x, agg, 0, nnWt, x, 1, nnWt, N, DD, DD, DD,
                                         nnb, 1, bn_g, bn_b, bn_m, bn_v);
  }

  (void)hipMemsetAsync(pooled, 0, (size_t)G * 256 * 4, stream);
  pool_kernel<<<(int)((ND + 255) / 256), 256, 0, stream>>>(x, batch, pooled, N);
  out_kernel<<<(G * 6 + 255) / 256, 256, 0, stream>>>(pooled, outW, outb, (float*)d_out, G);
}
